// GestureShallow_83958020702993
// MI455X (gfx1250) — matrix operand order repaired and checked by reading, not compiled, not run on hardware
//
#include <hip/hip_runtime.h>
#include <hip/hip_bf16.h>
#include <stddef.h>

// ---------------------------------------------------------------------------
// SLAYER SNN forward for MI455X (gfx1250, wave32).
//  - activations: time-major [t][n][c][h][w], spikes & pre-activations in f16
//  - convs / dense layers: implicit-GEMM via v_wmma_f32_16x16x32_f16
//  - conv input frames staged into LDS by the Tensor Data Mover (TDM) with
//    hardware row padding (D# pad_enable), waited via s_wait_tensorcnt
//  - psp_spike: exact f32 IIR recursion, one thread per neuron, serial over T,
//    with global_prefetch of upcoming timesteps
// ---------------------------------------------------------------------------

typedef __attribute__((ext_vector_type(16))) _Float16 v16h;
typedef __attribute__((ext_vector_type(8)))  float    v8f;
typedef __attribute__((ext_vector_type(4)))  unsigned int u32x4;
typedef __attribute__((ext_vector_type(8)))  int      i32x8;
typedef __attribute__((ext_vector_type(4)))  int      i32x4;

#define TT 300   // timesteps
#define NB 4     // batch

// ---- psp constants (fp32) -------------------------------------------------
#define A_SR 0.9048374180359595f    // exp(-1/10)
#define A_RF 0.3678794411714423f    // exp(-1)
#define C_SR 0.2718281828459045f    // e/10
#define C_RF (-54.3656365691809f)   // -2*10*e
#define THETA 10.0f

#if __has_builtin(__builtin_amdgcn_tensor_load_to_lds)
#define USE_TDM 1
#else
#define USE_TDM 0
#endif

// ---------------------------------------------------------------------------
__global__ void cvt_f32_f16(const float* __restrict__ a, _Float16* __restrict__ b, int n) {
  int i = blockIdx.x * blockDim.x + threadIdx.x;
  if (i < n) b[i] = (_Float16)a[i];
}

// 4x4 sum-pool of raw input (N,2,128,128,T) f32 -> time-major pre f16 [t][8192]
__global__ void pool4_in(const float* __restrict__ s_in, _Float16* __restrict__ pre) {
  const int j = blockIdx.x;                 // n*2048 + c*1024 + y*32 + x
  const int n = j >> 11, c = (j >> 10) & 1, y = (j >> 5) & 31, x = j & 31;
  const float* base = s_in + (size_t)(((n * 2 + c) * 128 + 4 * y) * 128 + 4 * x) * TT;
  for (int t = threadIdx.x; t < TT; t += blockDim.x) {
    float s = 0.f;
#pragma unroll
    for (int dy = 0; dy < 4; ++dy)
#pragma unroll
      for (int dx = 0; dx < 4; ++dx)
        s += base[(size_t)(dy * 128 + dx) * TT + t];
    pre[(size_t)t * 8192 + j] = (_Float16)(11.0f * s);   // 1.1*theta
  }
}

// 2x2 sum-pool on f16 spikes, time-major in/out.
__global__ void pool2_k(const _Float16* __restrict__ spk, _Float16* __restrict__ pre,
                        int C, int Hin, int Win) {
  const int Ho = Hin >> 1, Wo = Win >> 1;
  const int FO = NB * C * Ho * Wo;
  int idx = blockIdx.x * blockDim.x + threadIdx.x;
  if (idx >= FO * TT) return;
  int t = idx / FO, j = idx - t * FO;
  int n = j / (C * Ho * Wo); int r = j - n * (C * Ho * Wo);
  int c = r / (Ho * Wo);     int rr = r - c * (Ho * Wo);
  int y = rr / Wo, x = rr - y * Wo;
  const _Float16* f = spk + ((size_t)t * NB + n) * (size_t)(C * Hin * Win)
                          + (size_t)c * Hin * Win;
  float s = (float)f[(2 * y) * Win + 2 * x]     + (float)f[(2 * y) * Win + 2 * x + 1]
          + (float)f[(2 * y + 1) * Win + 2 * x] + (float)f[(2 * y + 1) * Win + 2 * x + 1];
  pre[idx] = (_Float16)(11.0f * s);
}

// psp_spike: exact f32 IIR + threshold. pre/spk time-major [t][F].
__global__ void psp_k(const _Float16* __restrict__ pre, _Float16* __restrict__ spk, int F) {
  int j = blockIdx.x * blockDim.x + threadIdx.x;
  if (j >= F) return;
  float gp = 0.f, hp = 0.f, gr = 0.f, hr = 0.f;
  for (int t = 0; t < TT; ++t) {
    // hide load latency along the serial chain (speculative, lands in WGP$)
    __builtin_prefetch(pre + (size_t)(t + 4) * F + j, 0, 1);
    float xi = (float)pre[(size_t)t * F + j];
    hp = A_SR * (hp + gp); gp = A_SR * gp + xi;
    hr = A_RF * (hr + gr); gr = A_RF * gr;
    float u = C_SR * hp + C_RF * hr;
    float s = (u >= THETA) ? 1.0f : 0.0f;
    spk[(size_t)t * F + j] = (_Float16)s;
    gr += s;
  }
}

// final psp over fc2 pre [t][n*16+oc] -> d_out (N,11,T) f32
__global__ void psp_final(const _Float16* __restrict__ pre, float* __restrict__ out) {
  int j = blockIdx.x * blockDim.x + threadIdx.x;
  if (j >= NB * 11) return;
  int n = j / 11, oc = j - n * 11;
  float gp = 0.f, hp = 0.f, gr = 0.f, hr = 0.f;
  for (int t = 0; t < TT; ++t) {
    float xi = (float)pre[(size_t)t * (NB * 16) + n * 16 + oc];
    hp = A_SR * (hp + gp); gp = A_SR * gp + xi;
    hr = A_RF * (hr + gr); gr = A_RF * gr;
    float u = C_SR * hp + C_RF * hr;
    float s = (u >= THETA) ? 1.0f : 0.0f;
    out[(size_t)(n * 11 + oc) * TT + t] = s;
    gr += s;
  }
}

// ---------------------------------------------------------------------------
// Implicit-GEMM conv: one workgroup per (t,n) frame. The raw frame (CIN*H rows
// of W f16) is DMA'd into LDS by the TDM with 2*PAD elements of row padding
// inserted by hardware (pad lands in pre-zeroed LDS => horizontal zero-pad is
// free). Rows live at stride Wp = W+2*PAD, data offset +PAD; vertical padding
// is a per-element guard. One wave per 16x16 output tile, K padded to 32.
// ---------------------------------------------------------------------------
template <int CIN, int H, int W, int KS, int PAD, int COUT>
__global__ void conv_wmma(const _Float16* __restrict__ spk,
                          const _Float16* __restrict__ wgt,
                          _Float16* __restrict__ pre) {
  constexpr int Wp = W + 2 * PAD;
  constexpr int ROWS = CIN * H;
  constexpr int Kreal = CIN * KS * KS;
  constexpr int nK = (Kreal + 31) / 32;
  constexpr int P = H * W;
  constexpr int TILES = (P / 16) * (COUT / 16);
  constexpr int LDSE = ROWS * Wp + 2 * PAD;
  __shared__ __align__(16) _Float16 lds[LDSE];

  const int frame = blockIdx.x;                       // t*NB + n
  const _Float16* in = spk + (size_t)frame * (CIN * H * W);
  _Float16* out = pre + (size_t)frame * (size_t)(COUT * P);

  // zero LDS (only the pad gaps strictly need it; blanket fill is cheap)
  for (int i = threadIdx.x; i < LDSE; i += blockDim.x) lds[i] = (_Float16)0.f;
  __syncthreads();

#if USE_TDM
  if (threadIdx.x < 32) {                             // wave 0 issues the DMA
    unsigned ldsb = (unsigned)(size_t)&lds[0] + 2u * PAD;   // +PAD elements
    unsigned long long ga = (unsigned long long)(size_t)in;
    u32x4 g0;
    g0[0] = 1u;                                       // count=1, user mode
    g0[1] = ldsb;                                     // lds_addr
    g0[2] = (unsigned)ga;                             // global_addr[31:0]
    g0[3] = (unsigned)((ga >> 32) & 0x01FFFFFFu) | 0x80000000u; // [56:32]|type=2
    constexpr unsigned IVL = (W == 32) ? 3u : (W == 16) ? 2u : 1u; // row dwords
    constexpr unsigned AMT = (unsigned)(PAD - 1);     // pad = PAD dwords
    i32x8 g1;
    g1[0] = (int)((1u << 16) | (1u << 20) | (IVL << 22) | (AMT << 25)); // 2B,pad
    g1[1] = (int)((unsigned)W << 16);                 // tensor_dim0[15:0]
    g1[2] = (int)((unsigned)ROWS << 16);              // tensor_dim1[15:0]
    g1[3] = (int)((unsigned)W << 16);                 // tile_dim0
    g1[4] = (int)ROWS;                                // tile_dim1
    g1[5] = (int)W;                                   // tensor_dim0_stride
    g1[6] = 0; g1[7] = 0;
    i32x4 z4 = {0, 0, 0, 0};
    i32x8 z8 = {0, 0, 0, 0, 0, 0, 0, 0};
    __builtin_amdgcn_tensor_load_to_lds(g0, g1, z4, z4, z8, 0);
    __builtin_amdgcn_s_wait_tensorcnt(0);
  }
#else
  for (int i = threadIdx.x; i < CIN * H * W; i += blockDim.x) {
    int r = i / W, x = i - r * W;
    lds[PAD + r * Wp + x] = in[i];
  }
#endif
  __syncthreads();

  const int wave = threadIdx.x >> 5;
  const int lane = threadIdx.x & 31;
  const int hf   = lane >> 4;          // lane half
  const int lid  = lane & 15;
  const int nw   = blockDim.x >> 5;

  for (int tile = wave; tile < TILES; tile += nw) {
    const int mt = tile % (COUT / 16), nt = tile / (COUT / 16);
    const int mBase = mt * 16, nBase = nt * 16;
    const int p = nBase + lid;
    const int oy = p / W, ox = p - oy * W;
    v8f acc = {};
    for (int ks = 0; ks < nK; ++ks) {
      v16h a, b;
#pragma unroll
      for (int e = 0; e < 16; ++e) {
        // A lane layout: K = e + 16*half (matches B)
        int ka = ks * 32 + e + (hf << 4);
        _Float16 av = (_Float16)0.f;
        if (ka < Kreal) av = wgt[(mBase + lid) * Kreal + ka];
        a[e] = av;
        // B lane layout: per-lane column, K = e + 16*half
        int kb = ks * 32 + e + (hf << 4);
        _Float16 bv = (_Float16)0.f;
        if (kb < Kreal) {
          int ic = kb / (KS * KS), rem = kb - ic * (KS * KS);
          int ky = rem / KS, kx = rem - ky * KS;
          int iy = oy + ky - PAD;                    // vertical pad guard
          if (iy >= 0 && iy < H)
            bv = lds[(ic * H + iy) * Wp + ox + kx];  // horiz pad: zeroed gaps
        }
        b[e] = bv;
      }
      acc = __builtin_amdgcn_wmma_f32_16x16x32_f16(false, a, false, b,
                                                   (short)0, acc, false, false);
    }
#pragma unroll
    for (int r = 0; r < 8; ++r) {
      int oc = mBase + r + (hf << 3);                // D: VGPR r -> M = r+8*half
      out[(size_t)oc * P + p] = (_Float16)acc[r];
    }
  }
}

// ---------------------------------------------------------------------------
// Dense layer: columns are (t,n) pairs, contiguous KD f16 spikes each.
// One wave per 16x16 tile.
// ---------------------------------------------------------------------------
template <int MROWS, int MREAL, int KD, int OCP>
__global__ void fc_wmma(const _Float16* __restrict__ spk,
                        const _Float16* __restrict__ wgt,
                        _Float16* __restrict__ pre) {
  constexpr int NC = NB * TT;
  constexpr int TILES = (MROWS / 16) * (NC / 16);
  const int wave = (blockIdx.x * blockDim.x + threadIdx.x) >> 5;
  if (wave >= TILES) return;
  const int lane = threadIdx.x & 31;
  const int hf = lane >> 4, lid = lane & 15;
  const int mt = wave % (MROWS / 16), nt = wave / (MROWS / 16);
  const int pair = nt * 16 + lid;                     // t*NB + n
  const _Float16* bcol = spk + (size_t)pair * KD;
  const int row = mt * 16 + lid;
  v8f acc = {};
  for (int ks = 0; ks < KD / 32; ++ks) {
    v16h a, b;
#pragma unroll
    for (int e = 0; e < 16; ++e) {
      int ka = ks * 32 + e + (hf << 4);
      a[e] = (row < MREAL) ? wgt[(size_t)row * KD + ka] : (_Float16)0.f;
      b[e] = bcol[ks * 32 + e + (hf << 4)];
    }
    acc = __builtin_amdgcn_wmma_f32_16x16x32_f16(false, a, false, b,
                                                 (short)0, acc, false, false);
  }
#pragma unroll
  for (int r = 0; r < 8; ++r) {
    int oc = mt * 16 + r + (hf << 3);
    pre[(size_t)pair * OCP + oc] = (_Float16)acc[r];
  }
}

// ---------------------------------------------------------------------------
extern "C" void kernel_launch(void* const* d_in, const int* in_sizes, int n_in,
                              void* d_out, int out_size, void* d_ws, size_t ws_size,
                              hipStream_t stream) {
  (void)in_sizes; (void)n_in; (void)out_size; (void)ws_size;
  const float* s_in = (const float*)d_in[0];
  const float* w1   = (const float*)d_in[1];
  const float* w2   = (const float*)d_in[2];
  const float* w3   = (const float*)d_in[3];
  const float* w4a  = (const float*)d_in[4];
  const float* w4b  = (const float*)d_in[5];
  float* out = (float*)d_out;
  char* ws = (char*)d_ws;

  size_t off = 0;
  auto alloc = [&](size_t bytes) {
    size_t o = off; off += (bytes + 255) & ~(size_t)255; return o;
  };
  // f16 weight copies
  size_t oW1  = alloc(1600u * 2);
  size_t oW2  = alloc(18432u * 2);
  size_t oW3  = alloc(36864u * 2);
  size_t oW4a = alloc((size_t)1048576 * 2);
  size_t oW4b = alloc(2816u * 2);
  // Region A: pre-activations (lifetimes disjoint), max = conv1 out 131072*300
  size_t oA = alloc((size_t)131072 * TT * 2);
  // Region B: even spike layers (max s2: 32768*300)
  size_t oB = alloc((size_t)32768 * TT * 2);
  // Region C: odd spike layers (max s1: 131072*300)
  size_t oC = alloc((size_t)131072 * TT * 2);

  _Float16* W1h  = (_Float16*)(ws + oW1);
  _Float16* W2h  = (_Float16*)(ws + oW2);
  _Float16* W3h  = (_Float16*)(ws + oW3);
  _Float16* W4ah = (_Float16*)(ws + oW4a);
  _Float16* W4bh = (_Float16*)(ws + oW4b);
  _Float16* A = (_Float16*)(ws + oA);
  _Float16* B = (_Float16*)(ws + oB);
  _Float16* C = (_Float16*)(ws + oC);

  cvt_f32_f16<<<(1600 + 255) / 256, 256, 0, stream>>>(w1, W1h, 1600);
  cvt_f32_f16<<<(18432 + 255) / 256, 256, 0, stream>>>(w2, W2h, 18432);
  cvt_f32_f16<<<(36864 + 255) / 256, 256, 0, stream>>>(w3, W3h, 36864);
  cvt_f32_f16<<<(1048576 + 255) / 256, 256, 0, stream>>>(w4a, W4ah, 1048576);
  cvt_f32_f16<<<(2816 + 255) / 256, 256, 0, stream>>>(w4b, W4bh, 2816);

  // layer 0: pool4 + psp  -> s0 (F=8192)
  pool4_in<<<8192, 128, 0, stream>>>(s_in, A);
  psp_k<<<8192 / 256, 256, 0, stream>>>(A, B, 8192);
  // conv1 (2->32, 5x5, pad2) + psp -> s1 (F=131072)
  conv_wmma<2, 32, 32, 5, 2, 32><<<TT * NB, 256, 0, stream>>>(B, W1h, A);
  psp_k<<<131072 / 256, 256, 0, stream>>>(A, C, 131072);
  // pool2 + psp -> s2 (F=32768)
  pool2_k<<<(32768 * TT) / 256, 256, 0, stream>>>(C, A, 32, 32, 32);
  psp_k<<<32768 / 256, 256, 0, stream>>>(A, B, 32768);
  // conv2 (32->64, 3x3, pad1) + psp -> s3 (F=65536)
  conv_wmma<32, 16, 16, 3, 1, 64><<<TT * NB, 256, 0, stream>>>(B, W2h, A);
  psp_k<<<65536 / 256, 256, 0, stream>>>(A, C, 65536);
  // pool2 + psp -> s4 (F=16384)
  pool2_k<<<(16384 * TT) / 256, 256, 0, stream>>>(C, A, 64, 16, 16);
  psp_k<<<16384 / 256, 256, 0, stream>>>(A, B, 16384);
  // conv3 (64->64, 3x3, pad1) + psp -> s5 (F=16384)
  conv_wmma<64, 8, 8, 3, 1, 64><<<TT * NB, 256, 0, stream>>>(B, W3h, A);
  psp_k<<<64, 256, 0, stream>>>(A, C, 16384);
  // fc1 (4096->256) + psp -> s6 (F=1024)
  fc_wmma<256, 256, 4096, 256><<<150, 256, 0, stream>>>(C, W4ah, A);
  psp_k<<<1024 / 256, 256, 0, stream>>>(A, B, 1024);
  // fc2 (256->11, padded to 16 rows) + final psp -> d_out (N,11,T) f32
  fc_wmma<16, 11, 256, 16><<<10, 256, 0, stream>>>(B, W4bh, A);
  psp_final<<<2, 32, 0, stream>>>(A, out);
}
